// MultiScaleEdgeConv_31473520345744
// MI455X (gfx1250) — compile-verified, matrix operand order checked by reading, not run on hardware
//
#include <hip/hip_runtime.h>
#include <hip/hip_bf16.h>
#include <math.h>

#define B_ 16
#define N_ 1024
#define K_ 16
#define EIN 132
#define KP1 160          // GEMM1 K padded to multiple of 32
#define ST1 168          // act1 LDS row stride (bf16), odd multiple of 16B
#define ST2 136          // act2 LDS row stride (bf16), odd multiple of 16B
#define EPSF 1e-8f
#define PI_F 3.14159265358979323846f

typedef __bf16 bf16;
typedef bf16 v16bf __attribute__((ext_vector_type(16)));
typedef float v8f __attribute__((ext_vector_type(8)));
typedef unsigned int v4u __attribute__((ext_vector_type(4)));
typedef unsigned int v8u __attribute__((ext_vector_type(8)));

__device__ __forceinline__ unsigned short f2bf(float f) {
  unsigned int u = __float_as_uint(f);
  u += 0x7FFFu + ((u >> 16) & 1u);      // round-to-nearest-even
  return (unsigned short)(u >> 16);
}

// Load one 16-element bf16 fragment row-chunk: elements [0..7] at p, [16..23] at p+16
__device__ __forceinline__ v16bf load_frag(const unsigned short* p) {
  v4u x = *reinterpret_cast<const v4u*>(p);
  v4u y = *reinterpret_cast<const v4u*>(p + 16);
  v8u w;
  w[0] = x[0]; w[1] = x[1]; w[2] = x[2]; w[3] = x[3];
  w[4] = y[0]; w[5] = y[1]; w[6] = y[2]; w[7] = y[3];
  return __builtin_bit_cast(v16bf, w);
}

__device__ __forceinline__ float wrap_dphi(float d) {
  float t = d + PI_F;
  t -= floorf(t * (0.5f / PI_F)) * (2.0f * PI_F);
  return t - PI_F;
}

// ---------------- kNN over (eta, phi) with phi wrap ----------------
__global__ void knn_kernel(const float* __restrict__ pts, int* __restrict__ idx) {
  int t = blockIdx.x * blockDim.x + threadIdx.x;
  if (t >= B_ * N_) return;
  int b = t / N_, i = t % N_;
  const float* eta = pts + (b * 2 + 0) * N_;
  const float* phi = pts + (b * 2 + 1) * N_;
  float ei = eta[i], pii = phi[i];
  float bestd[K_]; int besti[K_];
#pragma unroll
  for (int k = 0; k < K_; ++k) { bestd[k] = 3.4e38f; besti[k] = 0; }
  for (int j = 0; j < N_; ++j) {
    float de = ei - eta[j];
    float dp = wrap_dphi(pii - phi[j]);
    float d = de * de + dp * dp;
    if (d < bestd[K_ - 1]) {
      int p = K_ - 1;
      while (p > 0 && d < bestd[p - 1]) {
        bestd[p] = bestd[p - 1]; besti[p] = besti[p - 1]; --p;
      }
      bestd[p] = d; besti[p] = j;
    }
  }
  int* o = idx + t * K_;
#pragma unroll
  for (int k = 0; k < K_; ++k) o[k] = besti[k];
}

// ---------------- node = w_node @ relu(bn1(fts)) ----------------
__global__ void node_kernel(const float* __restrict__ fts, const float* __restrict__ s1,
                            const float* __restrict__ b1, const float* __restrict__ wn,
                            float* __restrict__ node) {
  int t = blockIdx.x * blockDim.x + threadIdx.x;
  if (t >= B_ * 64 * N_) return;
  int i = t % N_; int r = t / N_; int c = r % 64; int b = r / 64;
  float acc = 0.0f;
#pragma unroll 8
  for (int d = 0; d < 32; ++d) {
    float v = fts[(b * 32 + d) * N_ + i];
    v = fmaxf(v * s1[d] + b1[d], 0.0f);
    acc += wn[c * 32 + d] * v;
  }
  node[(b * 64 + c) * N_ + i] = acc;
}

// ---------------- per-point (pt, rapidity, phi) ----------------
__global__ void ppp_kernel(const float* __restrict__ lvs, float* __restrict__ ppp) {
  int t = blockIdx.x * blockDim.x + threadIdx.x;
  if (t >= B_ * N_) return;
  int b = t / N_, i = t % N_;
  float px = lvs[(b * 4 + 0) * N_ + i];
  float py = lvs[(b * 4 + 1) * N_ + i];
  float pz = lvs[(b * 4 + 2) * N_ + i];
  float e  = lvs[(b * 4 + 3) * N_ + i];
  ppp[(b * 3 + 0) * N_ + i] = sqrtf(px * px + py * py);
  ppp[(b * 3 + 1) * N_ + i] = 0.5f * log1pf(2.0f * pz / fmaxf(e - pz, 1e-20f));
  ppp[(b * 3 + 2) * N_ + i] = atan2f(py, px);
}

// ---------------- per-point valid sums + global count ----------------
__global__ void validsum_kernel(const unsigned char* __restrict__ mask,
                                const int* __restrict__ idx,
                                float* __restrict__ denom, float* __restrict__ cnt) {
  int t = blockIdx.x * blockDim.x + threadIdx.x;
  if (t >= B_ * N_) return;
  int b = t / N_, i = t % N_;
  float vi = mask[b * N_ + i] ? 1.0f : 0.0f;
  float s = 0.0f;
#pragma unroll
  for (int k = 0; k < K_; ++k) {
    int j = idx[(b * N_ + i) * K_ + k];
    s += vi * (mask[b * N_ + j] ? 1.0f : 0.0f);
  }
  denom[b * N_ + i] = fmaxf(s, 1.0f);
  atomicAdd(&cnt[b], s);
}

// ---------------- convert weights to bf16 (pad w_e1 K: 132 -> 160) ----------------
__global__ void prep_kernel(const float* __restrict__ w_e1, const float* __restrict__ w_e2,
                            unsigned short* __restrict__ w1bf, unsigned short* __restrict__ w2bf) {
  int stride = gridDim.x * blockDim.x;
  int t0 = blockIdx.x * blockDim.x + threadIdx.x;
  for (int x = t0; x < 128 * KP1; x += stride) {
    int m = x / KP1, k = x % KP1;
    w1bf[x] = (k < EIN) ? f2bf(w_e1[m * EIN + k]) : (unsigned short)0;
  }
  for (int x = t0; x < 128 * 128; x += stride) w2bf[x] = f2bf(w_e2[x]);
}

// ---------------- SE gate: y = sigmoid(relu(t @ w1^T) @ w2^T) ----------------
__global__ void se_kernel(const float* __restrict__ sums, const float* __restrict__ cnt,
                          const float* __restrict__ se_w1, const float* __restrict__ se_w2,
                          float* __restrict__ yb) {
  int b = blockIdx.x, c = threadIdx.x;
  __shared__ float tt[128];
  __shared__ float z[8];
  float n = cnt[b];
  n = (n == 0.0f) ? 1.0f : n;
  tt[c] = sums[b * 128 + c] / n;
  __syncthreads();
  if (c < 8) {
    float a = 0.0f;
    for (int d = 0; d < 128; ++d) a += tt[d] * se_w1[c * 128 + d];
    z[c] = fmaxf(a, 0.0f);
  }
  __syncthreads();
  float a = 0.0f;
#pragma unroll
  for (int m = 0; m < 8; ++m) a += z[m] * se_w2[c * 8 + m];
  yb[b * 128 + c] = 1.0f / (1.0f + expf(-a));
}

// ---------------- fused edge-conv (WMMA bf16), two passes ----------------
// mode 0: accumulate per-(b,c) masked channel sums (for SE)
// mode 1: recompute, scale by SE gate, aggregate over K, write output
__global__ __launch_bounds__(256) void edge_kernel(
    const int* __restrict__ idx, const float* __restrict__ node,
    const float* __restrict__ ppp, const float* __restrict__ lvs,
    const unsigned char* __restrict__ mask,
    const float* __restrict__ s2, const float* __restrict__ b2,
    const unsigned short* __restrict__ w1bf,
    const float* __restrict__ s3, const float* __restrict__ b3,
    const unsigned short* __restrict__ w2bf,
    float* __restrict__ sums,
    const float* __restrict__ yb, const float* __restrict__ denom,
    float* __restrict__ out, int mode) {
  __shared__ alignas(16) unsigned short act1[64 * ST1];  // 21.0 KB
  __shared__ alignas(16) unsigned short act2[64 * ST2];  // 17.0 KB
  __shared__ float svalid[64];

  const int tid = threadIdx.x;
  const int b = blockIdx.x >> 8;        // N_/4 = 256 tiles per batch
  const int itile = blockIdx.x & 255;
  const int i0 = itile * 4;

  __builtin_prefetch(w1bf, 0, 2);
  __builtin_prefetch(w2bf, 0, 2);

  // ---- stage 132-channel edge activations (relu(bn2(.)) -> bf16) ----
  {
    const int e = tid >> 2;             // 0..63 : edge = 16*i_local + k
    const int q = tid & 3;
    const int il = e >> 4;
    const int k = e & 15;
    const int i = i0 + il;
    const int j = idx[(b * N_ + i) * K_ + k];

    for (int c = q; c < 128; c += 4) {
      float v = (c < 64) ? node[(b * 64 + c) * N_ + i]
                         : node[(b * 64 + (c - 64)) * N_ + j];
      act1[e * ST1 + c] = f2bf(fmaxf(v * s2[c] + b2[c], 0.0f));
    }
    if (q == 0) {
      float pti = ppp[(b * 3 + 0) * N_ + i], rapi = ppp[(b * 3 + 1) * N_ + i];
      float phii = ppp[(b * 3 + 2) * N_ + i];
      float ptj = ppp[(b * 3 + 0) * N_ + j], rapj = ppp[(b * 3 + 1) * N_ + j];
      float phij = ppp[(b * 3 + 2) * N_ + j];
      float dp = wrap_dphi(phii - phij);
      float dr = rapi - rapj;
      float delta = sqrtf(dr * dr + dp * dp);
      float ptmin = fminf(pti, ptj);
      float px = lvs[(b * 4 + 0) * N_ + i] + lvs[(b * 4 + 0) * N_ + j];
      float py = lvs[(b * 4 + 1) * N_ + i] + lvs[(b * 4 + 1) * N_ + j];
      float pz = lvs[(b * 4 + 2) * N_ + i] + lvs[(b * 4 + 2) * N_ + j];
      float ee = lvs[(b * 4 + 3) * N_ + i] + lvs[(b * 4 + 3) * N_ + j];
      float m2 = fmaxf(ee * ee - px * px - py * py - pz * pz, EPSF);
      float f4[4];
      f4[0] = logf(fmaxf(ptmin * delta, EPSF));                       // lnkt
      f4[1] = logf(fmaxf(ptmin / fmaxf(pti + ptj, EPSF), EPSF));      // lnz
      f4[2] = logf(fmaxf(delta, EPSF));                               // lndelta
      f4[3] = logf(m2);                                               // lnm2
#pragma unroll
      for (int f = 0; f < 4; ++f) {
        int c = 128 + f;
        act1[e * ST1 + c] = f2bf(fmaxf(f4[f] * s2[c] + b2[c], 0.0f));
      }
      svalid[e] = (mask[b * N_ + i] && mask[b * N_ + j]) ? 1.0f : 0.0f;
    } else if (q == 1) {
      for (int c = EIN; c < KP1; ++c) act1[e * ST1 + c] = 0;  // zero pad K
    }
  }
  __syncthreads();

  const int lane = tid & 31;
  const int lr = lane & 15;
  const int half8 = (lane >> 4) << 3;   // 0 or 8 (K-offset within fragment)
  const int mtile = tid >> 5;           // wave id -> 16-channel M tile

  // ===== GEMM1: h = relu(bn3(w_e1 @ act1)) -> act2 (bf16 in LDS) =====
  v16bf a1[5];
  {
    const unsigned short* wrow = w1bf + (mtile * 16 + lr) * KP1;
#pragma unroll
    for (int s = 0; s < 5; ++s) a1[s] = load_frag(wrow + s * 32 + half8);
  }
#pragma unroll
  for (int nt = 0; nt < 4; ++nt) {
    const int e = nt * 16 + lr;
    const unsigned short* brow = act1 + e * ST1;
    v8f acc = {};
#pragma unroll
    for (int s = 0; s < 5; ++s) {
      v16bf bfr = load_frag(brow + s * 32 + half8);
      acc = __builtin_amdgcn_wmma_f32_16x16x32_bf16(false, a1[s], false, bfr,
                                                    (short)0, acc, false, false);
    }
    v4u pk;
#pragma unroll
    for (int t2 = 0; t2 < 4; ++t2) {
      int m0 = mtile * 16 + half8 + 2 * t2;
      float h0 = fmaxf(acc[2 * t2] * s3[m0] + b3[m0], 0.0f);
      float h1 = fmaxf(acc[2 * t2 + 1] * s3[m0 + 1] + b3[m0 + 1], 0.0f);
      pk[t2] = (unsigned int)f2bf(h0) | ((unsigned int)f2bf(h1) << 16);
    }
    *reinterpret_cast<v4u*>(act2 + e * ST2 + mtile * 16 + half8) = pk;
  }
  __syncthreads();

  // ===== GEMM2: x = w_e2 @ act2, then reduce =====
  v16bf a2[4];
  {
    const unsigned short* wrow = w2bf + (mtile * 16 + lr) * 128;
#pragma unroll
    for (int s = 0; s < 4; ++s) a2[s] = load_frag(wrow + s * 32 + half8);
  }
#pragma unroll
  for (int nt = 0; nt < 4; ++nt) {
    const int e = nt * 16 + lr;
    const unsigned short* brow = act2 + e * ST2;
    v8f acc = {};
#pragma unroll
    for (int s = 0; s < 4; ++s) {
      v16bf bfr = load_frag(brow + s * 32 + half8);
      acc = __builtin_amdgcn_wmma_f32_16x16x32_bf16(false, a2[s], false, bfr,
                                                    (short)0, acc, false, false);
    }
    const float v = svalid[e];
    if (mode == 0) {
#pragma unroll
      for (int r = 0; r < 8; ++r) {
        float s = acc[r] * v;           // mask each neighbor column
        s += __shfl_xor(s, 1, 32);      // sum over the 16 neighbors (lanes 0-15 / 16-31)
        s += __shfl_xor(s, 2, 32);
        s += __shfl_xor(s, 4, 32);
        s += __shfl_xor(s, 8, 32);
        if (lr == 0) atomicAdd(&sums[b * 128 + mtile * 16 + half8 + r], s);
      }
    } else {
      const int i = i0 + nt;            // this N-tile == all K neighbors of point i
      const float dn = denom[b * N_ + i];
#pragma unroll
      for (int r = 0; r < 8; ++r) {
        float s = acc[r] * v;
        s += __shfl_xor(s, 1, 32);
        s += __shfl_xor(s, 2, 32);
        s += __shfl_xor(s, 4, 32);
        s += __shfl_xor(s, 8, 32);
        if (lr == 0) {
          int c = mtile * 16 + half8 + r;
          out[(b * 128 + c) * N_ + i] = s * yb[b * 128 + c] / dn;
        }
      }
    }
  }
}

extern "C" void kernel_launch(void* const* d_in, const int* in_sizes, int n_in,
                              void* d_out, int out_size, void* d_ws, size_t ws_size,
                              hipStream_t stream) {
  (void)in_sizes; (void)n_in; (void)out_size; (void)ws_size;
  const float* pts   = (const float*)d_in[0];
  const float* fts   = (const float*)d_in[1];
  const float* lvs   = (const float*)d_in[2];
  const unsigned char* mask = (const unsigned char*)d_in[3];
  const float* bn1_s = (const float*)d_in[4];
  const float* bn1_b = (const float*)d_in[5];
  const float* wnode = (const float*)d_in[6];
  const float* bn2_s = (const float*)d_in[7];
  const float* bn2_b = (const float*)d_in[8];
  const float* w_e1  = (const float*)d_in[9];
  const float* bn3_s = (const float*)d_in[10];
  const float* bn3_b = (const float*)d_in[11];
  const float* w_e2  = (const float*)d_in[12];
  const float* se_w1 = (const float*)d_in[13];
  const float* se_w2 = (const float*)d_in[14];
  float* out = (float*)d_out;

  char* ws = (char*)d_ws;
  size_t off = 0;
  auto alloc = [&](size_t bytes) {
    void* p = ws + off;
    off = (off + bytes + 255) & ~(size_t)255;
    return p;
  };
  int*            idx   = (int*)alloc((size_t)B_ * N_ * K_ * 4);
  float*          node  = (float*)alloc((size_t)B_ * 64 * N_ * 4);
  float*          ppp   = (float*)alloc((size_t)B_ * 3 * N_ * 4);
  unsigned short* w1bf  = (unsigned short*)alloc(128 * KP1 * 2);
  unsigned short* w2bf  = (unsigned short*)alloc(128 * 128 * 2);
  float*          sums  = (float*)alloc((size_t)B_ * 128 * 4);
  float*          cnt   = (float*)alloc((size_t)B_ * 4);
  float*          yb    = (float*)alloc((size_t)B_ * 128 * 4);
  float*          denom = (float*)alloc((size_t)B_ * N_ * 4);

  hipMemsetAsync(sums, 0, (size_t)B_ * 128 * 4, stream);
  hipMemsetAsync(cnt, 0, (size_t)B_ * 4, stream);

  prep_kernel<<<64, 256, 0, stream>>>(w_e1, w_e2, w1bf, w2bf);
  knn_kernel<<<(B_ * N_ + 255) / 256, 256, 0, stream>>>(pts, idx);
  node_kernel<<<(B_ * 64 * N_ + 255) / 256, 256, 0, stream>>>(fts, bn1_s, bn1_b, wnode, node);
  ppp_kernel<<<(B_ * N_ + 255) / 256, 256, 0, stream>>>(lvs, ppp);
  validsum_kernel<<<(B_ * N_ + 255) / 256, 256, 0, stream>>>(mask, idx, denom, cnt);

  const int edge_blocks = B_ * (N_ / 4);  // 4096
  edge_kernel<<<edge_blocks, 256, 0, stream>>>(idx, node, ppp, lvs, mask, bn2_s, bn2_b,
                                               w1bf, bn3_s, bn3_b, w2bf, sums, yb, denom,
                                               out, 0);
  se_kernel<<<B_, 128, 0, stream>>>(sums, cnt, se_w1, se_w2, yb);
  edge_kernel<<<edge_blocks, 256, 0, stream>>>(idx, node, ppp, lvs, mask, bn2_s, bn2_b,
                                               w1bf, bn3_s, bn3_b, w2bf, sums, yb, denom,
                                               out, 1);
}
